// MPNNModel_48808008352181
// MI455X (gfx1250) — compile-verified, matrix operand order checked by reading, not run on hardware
//
#include <hip/hip_runtime.h>

// ---------- types ----------
typedef __attribute__((ext_vector_type(16))) __bf16 v16bf;
typedef __attribute__((ext_vector_type(8)))  float  v8f;
typedef __attribute__((ext_vector_type(4)))  unsigned int v4u;

static __device__ __forceinline__ unsigned short f2bf(float f) {
  // round-to-nearest-even fp32 -> bf16
  unsigned int u = __float_as_uint(f);
  u += 0x7FFFu + ((u >> 16) & 1u);
  return (unsigned short)(u >> 16);
}

// ---------------------------------------------------------------------------
// Pre-swizzle mpnn_W [30][256][256] f32 into WMMA-B-fragment-major bf16:
// Wswz[slice][ntile(16)][ktile(8)][lane(32)][h(16)]  with B[k][n] = W[n][k].
// 16-bit B layout (ISA 7.12.2): lane<16 covers k%32 in {0..7,16..23},
// lane>=16 covers +8; within a lane, VGPR r holds k-pair (2r,2r+1) pattern.
// ---------------------------------------------------------------------------
__global__ __launch_bounds__(256) void swizzle_weights_kernel(
    const float* __restrict__ W, unsigned short* __restrict__ Wswz) {
  long idx = (long)blockIdx.x * 256 + threadIdx.x;   // 30*65536 total
  if (idx >= 30L * 65536L) return;
  int slice = (int)(idx >> 16);
  int rem   = (int)(idx & 65535);
  int h     = rem & 15;
  int lane  = (rem >> 4) & 31;
  int kt    = (rem >> 9) & 7;
  int ntile = (rem >> 12) & 15;
  int n = ntile * 16 + (lane & 15);
  int r = h >> 1, par = h & 1;
  int kof = (r < 4) ? (2 * r) : (16 + 2 * (r - 4));
  int k = kt * 32 + kof + ((lane >> 4) << 3) + par;
  Wswz[idx] = f2bf(W[((long)slice * 256 + n) * 256 + k]);
}

// ---------------------------------------------------------------------------
// Encoder: one block per node, thread t computes channel t of
//   x1 @ W1.T + b1 + x2 @ W2.T + b2   -> bf16 node state
// ---------------------------------------------------------------------------
__global__ __launch_bounds__(256) void encode_kernel(
    const float* __restrict__ x1, int d1, const float* __restrict__ W1, const float* __restrict__ b1,
    const float* __restrict__ x2, int d2, const float* __restrict__ W2, const float* __restrict__ b2,
    unsigned short* __restrict__ xs_bf, int nrows, int row_off) {
  __shared__ float sx[192];
  int row = blockIdx.x;
  if (row >= nrows) return;
  int tid = threadIdx.x;
  if (tid < d1)            sx[tid] = x1[(long)row * d1 + tid];
  else if (tid < d1 + d2)  sx[tid] = x2[(long)row * d2 + (tid - d1)];
  __syncthreads();
  float acc = b1[tid] + b2[tid];
  const float* w1r = W1 + (long)tid * d1;
  for (int k = 0; k < d1; ++k) acc += sx[k] * w1r[k];
  const float* w2r = W2 + (long)tid * d2;
  for (int k = 0; k < d2; ++k) acc += sx[d1 + k] * w2r[k];
  xs_bf[(long)(row_off + row) * 256 + tid] = f2bf(acc);
}

// pt/ft rows: broadcast embeddings; boundary n_pt read from device scalar.
__global__ __launch_bounds__(256) void fill_emb_kernel(
    const float* __restrict__ emb_pt, const float* __restrict__ emb_ft,
    unsigned short* __restrict__ xs_bf, int row_off, int nrows,
    const int* __restrict__ d_npt) {
  int npt = *d_npt;
  long idx = (long)blockIdx.x * 256 + threadIdx.x;
  if (idx >= (long)nrows * 256) return;
  int row = (int)(idx >> 8);
  int ch  = (int)(idx & 255);
  float v = (row < npt) ? emb_pt[ch] : emb_ft[ch];
  xs_bf[(long)row_off * 256 + idx] = f2bf(v);
}

__global__ __launch_bounds__(256) void zero_kernel(float4* __restrict__ p, long n4) {
  long i = (long)blockIdx.x * 256 + threadIdx.x;
  if (i < n4) p[i] = make_float4(0.f, 0.f, 0.f, 0.f);
}

__global__ __launch_bounds__(256) void relu_store_kernel(
    const float* __restrict__ acc, unsigned short* __restrict__ xs_bf,
    float* __restrict__ fin, long n) {
  long i = (long)blockIdx.x * 256 + threadIdx.x;
  if (i >= n) return;
  float v = acc[i];
  v = v > 0.f ? v : 0.f;
  xs_bf[i] = f2bf(v);
  if (fin) fin[i] = v;
}

// ---------------------------------------------------------------------------
// Edge GEMM: block = 8 wave32s, 32 edges. Gather bf16 rows into LDS in
// A-fragment-swizzled layout, run v_wmma_f32_16x16x32_bf16 over K=256 with
// two interleaved independent accumulator chains per pass, epilogue =
// +bias, ReLU, *w[e], branchless atomicAdd scatter (L2-resident target).
// Padded edges (e >= E) gather node 0 with w = 0 -> scatter adds 0.0f.
// ---------------------------------------------------------------------------
#define TM 32
__global__ __launch_bounds__(256) void edge_mpnn_kernel(
    const unsigned short* __restrict__ xs_bf,   // [total,256] bf16 node states
    const unsigned short* __restrict__ Wswz,    // swizzled W.T slice (65536 halves)
    const float* __restrict__ bias,             // [256]
    const int*  __restrict__ src,               // [E]
    const int*  __restrict__ tgt,               // [E]
    const float* __restrict__ ew,               // [E]
    float* __restrict__ out,                    // [total,256] f32 accumulator
    int E, int src_base, int tgt_base, int src_addn, int tgt_addn,
    const int* __restrict__ d_npt) {
  __shared__ __align__(32) unsigned short Asw[2 * 8 * 32 * 16];  // 16 KB
  __shared__ int   s_src[TM];
  __shared__ int   s_tgt[TM];
  __shared__ float s_w[TM];

  int npt  = *d_npt;
  int soff = src_base + (src_addn ? npt : 0);
  int toff = tgt_base + (tgt_addn ? npt : 0);
  int e0   = blockIdx.x * TM;
  int tid  = threadIdx.x;

  if (tid < TM) {
    int e = e0 + tid;
    int si = 0, ti = 0; float wv = 0.f;   // padded edges: node 0, weight 0
    if (e < E) { si = src[e]; ti = tgt[e]; wv = ew[e]; }
    s_src[tid] = si; s_tgt[tid] = ti; s_w[tid] = wv;
  }
  __syncthreads();

  // Stage A into LDS, swizzled: each 16B global chunk -> one 16B LDS chunk.
  // chunk c: row = c/32, k = (c%32)*8  (8 bf16 = 16 bytes)
  for (int c = tid; c < TM * 32; c += 256) {
    int row = c >> 5;
    int k   = (c & 31) << 3;
    v4u v = *(const v4u*)(xs_bf + (long)(soff + s_src[row]) * 256 + k);
    int kk    = k & 31;
    int ktile = k >> 5;
    int hi    = (kk >> 3) & 1;          // lane-half select
    int h0    = (kk & 16) ? 8 : 0;      // half offset within lane's 16
    int lane  = (row & 15) + (hi << 4);
    int mt    = row >> 4;
    *(v4u*)&Asw[(((mt * 8 + ktile) * 32 + lane) << 4) + h0] = v;
  }
  __syncthreads();

  int lane = tid & 31;
  int wid  = tid >> 5;          // 8 waves
  int mt   = wid >> 2;          // M tile (0..1) -> edges mt*16..mt*16+15
  int nb   = wid & 3;           // N-tile phase

  // A fragments (reused across all 4 N-tiles)
  v16bf a[8];
#pragma unroll
  for (int kt = 0; kt < 8; ++kt)
    a[kt] = *(const v16bf*)&Asw[((mt * 8 + kt) * 32 + lane) << 4];

  // C/D layout: lane<16: m = v, n = lane; lane>=16: m = v+8, n = lane-16
  int nloc  = lane & 15;
  int mbase = (lane >> 4) << 3;
  int row0  = mt * 16 + mbase;
  int   tgts[8];
  float wvs[8];
#pragma unroll
  for (int v = 0; v < 8; ++v) { tgts[v] = s_tgt[row0 + v]; wvs[v] = s_w[row0 + v]; }

  float* outp = out + (long)toff * 256;

#pragma unroll
  for (int np = 0; np < 2; ++np) {
    // two independent accumulator chains, interleaved for matrix-pipe ILP
    int ntA = nb + np * 4;        // covers ntiles 0..7  across nb,np
    int ntB = ntA + 8;            // covers ntiles 8..15
    const unsigned short* wpA = Wswz + (((long)ntA * 8) * 32 + lane) * 16;
    const unsigned short* wpB = Wswz + (((long)ntB * 8) * 32 + lane) * 16;
    v8f cA = {}, cB = {};
#pragma unroll
    for (int kt = 0; kt < 8; ++kt) {
      v16bf bA = *(const v16bf*)(wpA + (long)kt * 32 * 16);
      v16bf bB = *(const v16bf*)(wpB + (long)kt * 32 * 16);
      cA = __builtin_amdgcn_wmma_f32_16x16x32_bf16(
               false, a[kt], false, bA, (short)0, cA, false, false);
      cB = __builtin_amdgcn_wmma_f32_16x16x32_bf16(
               false, a[kt], false, bB, (short)0, cB, false, false);
    }
    int   nA  = ntA * 16 + nloc;
    int   nB  = ntB * 16 + nloc;
    float bnA = bias[nA];
    float bnB = bias[nB];
#pragma unroll
    for (int v = 0; v < 8; ++v) {
      float vA = cA[v] + bnA;
      vA = vA > 0.f ? vA : 0.f;
      vA *= wvs[v];
      float vB = cB[v] + bnB;
      vB = vB > 0.f ? vB : 0.f;
      vB *= wvs[v];
      long rbase = (long)tgts[v] * 256;
      atomicAdd(&outp[rbase + nA], vA);
      atomicAdd(&outp[rbase + nB], vB);
    }
  }
}

// ---------------------------------------------------------------------------
static inline size_t alup(size_t x) { return (x + 255) & ~(size_t)255; }

extern "C" void kernel_launch(void* const* d_in, const int* in_sizes, int n_in,
                              void* d_out, int out_size, void* d_ws, size_t ws_size,
                              hipStream_t stream) {
  const float* x_cl   = (const float*)d_in[0];
  const float* x_cc   = (const float*)d_in[1];
  const float* x_al   = (const float*)d_in[2];
  const float* x_ac   = (const float*)d_in[3];
  const float* W_cl   = (const float*)d_in[4];
  const float* b_cl   = (const float*)d_in[5];
  const float* W_cc   = (const float*)d_in[6];
  const float* b_cc   = (const float*)d_in[7];
  const float* W_al   = (const float*)d_in[8];
  const float* b_al   = (const float*)d_in[9];
  const float* W_ac   = (const float*)d_in[10];
  const float* b_ac   = (const float*)d_in[11];
  const float* emb_pt = (const float*)d_in[12];
  const float* emb_ft = (const float*)d_in[13];
  const float* mpnn_W = (const float*)d_in[14];
  const float* mpnn_b = (const float*)d_in[15];
  const float* w_e    = (const float*)d_in[16];
  const int*   es     = (const int*)d_in[17];
  const int*   d_npt  = (const int*)d_in[18];

  int n_cl  = in_sizes[0] / 128;
  int n_ac  = in_sizes[2] / 96;
  int total = out_size / 256;
  int E     = in_sizes[17] / 12;        // es: [6][2][E]
  int n_ptft = total - n_cl - n_ac;

  // workspace: f32 accumulator | bf16 node states | swizzled bf16 weights
  size_t off = 0;
  float*          out_acc = (float*)((char*)d_ws + off);          off += alup((size_t)total * 256 * 4);
  unsigned short* xs_bf   = (unsigned short*)((char*)d_ws + off); off += alup((size_t)total * 256 * 2);
  unsigned short* Wswz    = (unsigned short*)((char*)d_ws + off);

  // 1) pre-swizzle weights to WMMA fragment layout (bf16)
  swizzle_weights_kernel<<<30 * 256, 256, 0, stream>>>(mpnn_W, Wswz);

  // 2) encoders
  encode_kernel<<<n_cl, 256, 0, stream>>>(x_cl, 128, W_cl, b_cl,
                                          x_cc, 64,  W_cc, b_cc, xs_bf, n_cl, 0);
  encode_kernel<<<n_ac, 256, 0, stream>>>(x_al, 96,  W_al, b_al,
                                          x_ac, 32,  W_ac, b_ac, xs_bf, n_ac, n_cl);
  {
    long elems = (long)n_ptft * 256;
    fill_emb_kernel<<<(int)((elems + 255) / 256), 256, 0, stream>>>(
        emb_pt, emb_ft, xs_bf, n_cl + n_ac, n_ptft, d_npt);
  }

  static const int T_SRCS[6] = {0, 1, 2, 3, 1, 0};
  static const int T_TGTS[6] = {1, 0, 3, 2, 2, 3};
  int base[4] = {0, n_cl, n_cl + n_ac, n_cl + n_ac};  // type 3 adds *d_npt in-kernel
  long n_elems = (long)total * 256;

  for (int l = 0; l < 5; ++l) {
    zero_kernel<<<(int)((n_elems / 4 + 255) / 256), 256, 0, stream>>>(
        (float4*)out_acc, n_elems / 4);
    for (int j = 0; j < 6; ++j) {
      int s = T_SRCS[j], t = T_TGTS[j];
      edge_mpnn_kernel<<<(E + TM - 1) / TM, 256, 0, stream>>>(
          xs_bf,
          Wswz + (size_t)(l * 6 + j) * 65536,
          mpnn_b + (size_t)(l * 6 + j) * 256,
          es + (size_t)j * 2 * E,
          es + (size_t)j * 2 * E + E,
          w_e + (size_t)j * E,
          out_acc, E,
          base[s], base[t], (s == 3) ? 1 : 0, (t == 3) ? 1 : 0, d_npt);
    }
    relu_store_kernel<<<(int)((n_elems + 255) / 256), 256, 0, stream>>>(
        out_acc, xs_bf, (l == 4) ? (float*)d_out : (float*)nullptr, n_elems);
  }
}
